// BatchMixingLoss_27084063769196
// MI455X (gfx1250) — matrix operand order repaired and checked by reading, not compiled, not run on hardware
//
#include <hip/hip_runtime.h>
#include <hip/hip_bf16.h>

// ---------------------------------------------------------------------------
// BatchMixingLoss for MI455X (gfx1250, wave32, WMMA).
// Fused 3-kernel pipeline; D (256MB) is never materialized: recomputed from an
// L2-resident bf16 copy of X (8MB) with v_wmma_f32_16x16x32_bf16.
// ---------------------------------------------------------------------------

typedef __attribute__((ext_vector_type(16))) __bf16 v16bf;
typedef __attribute__((ext_vector_type(8)))  __bf16 v8bf;
typedef __attribute__((ext_vector_type(8)))  float  v8f;

#define N_CELLS   8192
#define DIMS      512
#define NTILE     (N_CELLS / 16)   // 512 column tiles of 16
#define KCHUNKS   (DIMS / 32)      // 16 WMMA K-steps of 32
#define NWAVES    4                // 128 threads / 32
#define SENTINEL  1.0e10f
#define EPSV      1e-8f

// ---------------- Kernel 1: sq-norms + f32 -> bf16 conversion --------------
__global__ __launch_bounds__(128)
void prep_kernel(const float* __restrict__ X, __bf16* __restrict__ Xbf,
                 float* __restrict__ sqn) {
  __shared__ float red[128];
  const int row = blockIdx.x;
  const float* xr = X + (size_t)row * DIMS;
  __bf16* br = Xbf + (size_t)row * DIMS;
  float s = 0.f;
  for (int t = threadIdx.x; t < DIMS; t += 128) {
    float v = xr[t];
    s += v * v;
    br[t] = (__bf16)v;
  }
  red[threadIdx.x] = s;
  __syncthreads();
  for (int off = 64; off > 0; off >>= 1) {
    if (threadIdx.x < off) red[threadIdx.x] += red[threadIdx.x + off];
    __syncthreads();
  }
  if (threadIdx.x == 0) sqn[row] = red[0];
}

// ---------------- Kernel 2: per-16-row strip, two WMMA passes --------------
__global__ __launch_bounds__(128)
void strip_kernel(const __bf16* __restrict__ Xbf, const float* __restrict__ sqn,
                  const int* __restrict__ labels, float* __restrict__ entOut) {
  __shared__ float tileLds[NWAVES][16][17];      // D-tile transpose buffer
  __shared__ float topLds[NWAVES][16][15];       // per-wave top-15 per row
  __shared__ float rowDmin[16];
  __shared__ float rowKth[16];
  __shared__ float statLds[NWAVES][32][8][5];    // Z, T, M0, M1, M2 partials

  const int lane = threadIdx.x & 31;
  const int wave = threadIdx.x >> 5;
  const int half = lane >> 4;        // which 16-lane half
  const int nloc = lane & 15;        // tile column / A row owned by this lane
  const int i0   = blockIdx.x * 16;

  // ---- A fragments (16 rows x 512) resident in VGPRs for both passes ----
  // A 16x32 layout (matches B): lane holds row M=nloc, K = 16*half..+15
  // -> 16 contiguous bf16.
  v16bf afrag[KCHUNKS];
  {
    const __bf16* arow = Xbf + (size_t)(i0 + nloc) * DIMS + 16 * half;
#pragma unroll
    for (int kc = 0; kc < KCHUNKS; ++kc) {
      afrag[kc] = *(const v16bf*)(arow + kc * 32);
    }
  }

  // squared norms of the 8 C rows (M = v + 8*half) this lane covers
  float sqnI[8];
#pragma unroll
  for (int v = 0; v < 8; ++v) sqnI[v] = sqn[i0 + v + 8 * half];

  // =================== PASS A: per-row top-15 distances ===================
  float top15[15];
#pragma unroll
  for (int t = 0; t < 15; ++t) top15[t] = 3.0e38f;

  for (int jt = wave; jt < NTILE; jt += NWAVES) {
    const int j0 = jt * 16;
    // B 32x16 operand, same contiguous k order as A: lane holds col N=nloc, K = 16*half..+15
    // -> 16 contiguous bf16 from row (j0+nloc) of X.
    const __bf16* brow = Xbf + (size_t)(j0 + nloc) * DIMS + 16 * half;
    const int jn = jt + NWAVES;
    if (jn < NTILE)
      __builtin_prefetch(Xbf + (size_t)(jn * 16 + nloc) * DIMS + 16 * half, 0, 3);

    v8f c = {};
#pragma unroll
    for (int kc = 0; kc < KCHUNKS; ++kc) {
      v16bf b = *(const v16bf*)(brow + kc * 32);
      c = __builtin_amdgcn_wmma_f32_16x16x32_bf16(
              false, afrag[kc], false, b, (short)0, c, false, false);
    }

    const float sqj = sqn[j0 + nloc];
#pragma unroll
    for (int v = 0; v < 8; ++v) {
      float d = sqnI[v] + sqj - 2.0f * c[v];
      if (i0 + v + 8 * half == j0 + nloc) d += SENTINEL;   // diagonal mask
      tileLds[wave][v + 8 * half][nloc] = d;
    }
    // transpose-consume: lanes 0..15 each own one row, update sorted top-15
    if (lane < 16) {
#pragma unroll
      for (int n2 = 0; n2 < 16; ++n2) {
        float val = tileLds[wave][lane][n2];
        if (val < top15[14]) {
          top15[14] = val;                 // replace max, bubble toward front
#pragma unroll
          for (int t = 14; t > 0; --t) {
            float a = top15[t - 1], b2 = top15[t];
            top15[t - 1] = fminf(a, b2);
            top15[t]     = fmaxf(a, b2);
          }
        }
      }
    }
  }

  // merge the 4 per-wave top-15 lists -> global Dmin and 15th-smallest
  if (lane < 16) {
#pragma unroll
    for (int t = 0; t < 15; ++t) topLds[wave][lane][t] = top15[t];
  }
  __syncthreads();
  if (wave == 0 && lane < 16) {
    float m15[15];
#pragma unroll
    for (int t = 0; t < 15; ++t) m15[t] = 3.0e38f;
    for (int w = 0; w < NWAVES; ++w) {
#pragma unroll
      for (int t = 0; t < 15; ++t) {
        float val = topLds[w][lane][t];
        if (val < m15[14]) {
          m15[14] = val;
#pragma unroll
          for (int q = 14; q > 0; --q) {
            float a = m15[q - 1], b2 = m15[q];
            m15[q - 1] = fminf(a, b2);
            m15[q]     = fmaxf(a, b2);
          }
        }
      }
    }
    rowDmin[lane] = m15[0];
    rowKth[lane]  = m15[14];
  }
  __syncthreads();

  // =========== PASS B: softmax + k-mask + batch-distribution sums =========
  float dminR[8], kthR[8];
#pragma unroll
  for (int v = 0; v < 8; ++v) {
    dminR[v] = rowDmin[v + 8 * half];
    kthR[v]  = rowKth[v + 8 * half];
  }
  float accZ[8], accT[8], accM0[8], accM1[8], accM2[8];
#pragma unroll
  for (int v = 0; v < 8; ++v) {
    accZ[v] = 0.f; accT[v] = 0.f; accM0[v] = 0.f; accM1[v] = 0.f; accM2[v] = 0.f;
  }

  for (int jt = wave; jt < NTILE; jt += NWAVES) {
    const int j0 = jt * 16;
    const __bf16* brow = Xbf + (size_t)(j0 + nloc) * DIMS + 16 * half;
    const int jn = jt + NWAVES;
    if (jn < NTILE)
      __builtin_prefetch(Xbf + (size_t)(jn * 16 + nloc) * DIMS + 16 * half, 0, 3);

    v8f c = {};
#pragma unroll
    for (int kc = 0; kc < KCHUNKS; ++kc) {
      v16bf b = *(const v16bf*)(brow + kc * 32);
      c = __builtin_amdgcn_wmma_f32_16x16x32_bf16(
              false, afrag[kc], false, b, (short)0, c, false, false);
    }

    const float sqj = sqn[j0 + nloc];
    const int   lab = labels[j0 + nloc];
    const float isb0 = (lab == 0) ? 1.f : 0.f;
    const float isb1 = (lab == 1) ? 1.f : 0.f;
    const float isb2 = (lab == 2) ? 1.f : 0.f;
#pragma unroll
    for (int v = 0; v < 8; ++v) {
      float d = sqnI[v] + sqj - 2.0f * c[v];
      if (i0 + v + 8 * half == j0 + nloc) d += SENTINEL;
      float e  = __expf(dminR[v] - d);                  // softmax numerator
      float sg = 1.0f / (1.0f + __expf(d - kthR[v]));   // sigmoid(kth - d)
      float es = e * sg;
      accZ[v] += e;  accT[v] += es;
      accM0[v] += es * isb0;  accM1[v] += es * isb1;  accM2[v] += es * isb2;
    }
  }

  // deterministic cross-lane / cross-wave reduction via LDS
#pragma unroll
  for (int v = 0; v < 8; ++v) {
    statLds[wave][lane][v][0] = accZ[v];
    statLds[wave][lane][v][1] = accT[v];
    statLds[wave][lane][v][2] = accM0[v];
    statLds[wave][lane][v][3] = accM1[v];
    statLds[wave][lane][v][4] = accM2[v];
  }
  __syncthreads();
  if (wave == 0 && lane < 16) {
    const int h = lane >> 3, v = lane & 7;   // row m=lane -> half, vgpr slot
    float Z = 0.f, T = 0.f, M0 = 0.f, M1 = 0.f, M2 = 0.f;
    for (int w = 0; w < NWAVES; ++w) {
      for (int lg = 0; lg < 16; ++lg) {
        const int src = h * 16 + lg;
        Z  += statLds[w][src][v][0];
        T  += statLds[w][src][v][1];
        M0 += statLds[w][src][v][2];
        M1 += statLds[w][src][v][3];
        M2 += statLds[w][src][v][4];
      }
    }
    // w_j = (E_j/Z)*s_j / (T/Z + EPS)  =>  p_b = M_b / (T + EPS*Z)
    const float denom = T + EPSV * Z;
    const float p0 = M0 / denom, p1 = M1 / denom, p2 = M2 / denom;
    const float ent = -(p0 * __logf(p0 + EPSV) +
                        p1 * __logf(p1 + EPSV) +
                        p2 * __logf(p2 + EPSV));
    entOut[i0 + lane] = ent / (__logf(3.0f) + EPSV);
  }
}

// ---------------- Kernel 3: deterministic mean reduction -------------------
__global__ __launch_bounds__(256)
void reduce_kernel(const float* __restrict__ ent, float* __restrict__ out) {
  __shared__ float red[256];
  float s = 0.f;
  for (int t = threadIdx.x; t < N_CELLS; t += 256) s += ent[t];
  red[threadIdx.x] = s;
  __syncthreads();
  for (int off = 128; off > 0; off >>= 1) {
    if (threadIdx.x < off) red[threadIdx.x] += red[threadIdx.x + off];
    __syncthreads();
  }
  if (threadIdx.x == 0) out[0] = -(red[0] / (float)N_CELLS);
}

// ---------------------------------------------------------------------------
extern "C" void kernel_launch(void* const* d_in, const int* in_sizes, int n_in,
                              void* d_out, int out_size, void* d_ws, size_t ws_size,
                              hipStream_t stream) {
  const float* emb    = (const float*)d_in[0];   // (8192, 512) f32
  const int*   labels = (const int*)d_in[1];     // (8192,) i32
  float*       out    = (float*)d_out;           // scalar

  // workspace layout: bf16 X copy (8MB) | sqnorms (32KB) | row entropies (32KB)
  __bf16* Xbf = (__bf16*)d_ws;
  float*  sqn = (float*)((char*)d_ws + (size_t)N_CELLS * DIMS * sizeof(__bf16));
  float*  ent = sqn + N_CELLS;

  prep_kernel  <<<N_CELLS,      128, 0, stream>>>(emb, Xbf, sqn);
  strip_kernel <<<N_CELLS / 16, 128, 0, stream>>>(Xbf, sqn, labels, ent);
  reduce_kernel<<<1,            256, 0, stream>>>(ent, out);
}
